// LocationAwareAttention_61366492725637
// MI455X (gfx1250) — compile-verified, matrix operand order checked by reading, not run on hardware
//
#include <hip/hip_runtime.h>
#include <hip/hip_bf16.h>
#include <math.h>

// MI455X / gfx1250, wave32. Matrix work via v_wmma_f32_16x16x32_bf16.

typedef __attribute__((ext_vector_type(16))) __bf16 v16bf;
typedef __attribute__((ext_vector_type(8)))  float  v8f;

#define B_   32
#define VL_  2048
#define H_   512

// f32 -> bf16 pair, truncation, single v_perm_b32
__device__ __forceinline__ unsigned int pack_bf16_fast(float x, float y) {
  union { float f; unsigned int u; } a, b;
  a.f = x; b.f = y;
#if __has_builtin(__builtin_amdgcn_perm)
  // result bytes: [0,1] = x bytes [2,3] (src1), [2,3] = y bytes [2,3] (src0)
  return __builtin_amdgcn_perm(b.u, a.u, 0x07060302u);
#else
  return (a.u >> 16) | (b.u & 0xFFFF0000u);
#endif
}

// ---------------------------------------------------------------------------
// Kernel 1: qcomb[b,d] = query[b]·Wq[d] + bias[d] + conv_b[d];  Wv -> bf16 (RNE)
// ---------------------------------------------------------------------------
__global__ __launch_bounds__(256)
void prep_kernel(const float* __restrict__ query,
                 const float* __restrict__ Wq,
                 const float* __restrict__ Wv,
                 const float* __restrict__ conv_b,
                 const float* __restrict__ bias,
                 float* __restrict__ qcomb,
                 unsigned short* __restrict__ wvbf) {
  const int tid = blockIdx.x * blockDim.x + threadIdx.x;   // 16384 threads
  {
    const int b = tid >> 9, d = tid & 511;
    const float* qrow = query + b * H_;
    const float* wrow = Wq + d * H_;
    float acc = 0.f;
    #pragma unroll 4
    for (int h = 0; h < H_; ++h) acc += qrow[h] * wrow[h];
    qcomb[tid] = acc + bias[d] + conv_b[d];
  }
  for (int i = tid; i < H_ * H_; i += 16384) {
    union { float f; unsigned int u; } v; v.f = Wv[i];
    unsigned int r = v.u + 0x7FFFu + ((v.u >> 16) & 1u);
    wvbf[i] = (unsigned short)(r >> 16);
  }
}

// ---------------------------------------------------------------------------
// Kernel 2: fused  v = value@Wv^T  ->  tanh(v+qcomb+conv)  ->  score_w dot
//           -> sigmoid -> scores[b,t].
// Grid: B * (VL/32) blocks of 256 threads (8 waves).
// Wave (tg,dg): 16 t-rows (tg) x 128 d-cols (dg), 8 WMMA accumulators.
// ---------------------------------------------------------------------------
__global__ __launch_bounds__(256)
void score_kernel(const float* __restrict__ value,
                  const float* __restrict__ last_attn,
                  const float* __restrict__ conv_w,    // (H,1,3)
                  const float* __restrict__ score_w,   // (1,H)
                  const float* __restrict__ score_b,   // (1,)
                  const float* __restrict__ qcomb,     // (B,H)
                  const unsigned short* __restrict__ wvbf, // (H,H) bf16
                  float* __restrict__ scores) {        // (B,VL)
  __shared__ float sLA[36];          // last_attn window with halo
  __shared__ float sPart[32 * 4];    // per-trow per-dgroup partial scores

  const int b    = blockIdx.x >> 6;
  const int t0b  = (blockIdx.x & 63) * 32;
  const int lane = threadIdx.x & 31;
  const int wave = threadIdx.x >> 5;
  const int tg   = wave & 1;
  const int dg   = wave >> 1;
  const int r    = lane & 15;        // A: row M / B: col N / C: col N
  const int hi   = lane >> 4;        // selects K half (A/B), row half (C)
  const int t0   = t0b + tg * 16;
  const int d0   = dg * 128;

  if (threadIdx.x < 34) {
    const int tt = t0b - 1 + (int)threadIdx.x;
    sLA[threadIdx.x] = (tt >= 0 && tt < VL_) ? last_attn[b * VL_ + tt] : 0.f;
  }
  __syncthreads();

  // ISA 7.12.2 16-bit fragment layout: K pairs at ((j&4)<<2)+2(j&3)+8*hi.
  // j=0..3 pairs are contiguous (k0+8hi .. +7), j=4..7 at +16: b128-friendly.
  const int khalf = 8 * hi;

  const float* aRow = value + (size_t)(b * VL_ + t0 + r) * H_;
  const unsigned short* bBase = wvbf + (size_t)(d0 + r) * H_;

  union AF { unsigned int u[8]; uint4 u4[2]; v16bf v; };
  const v8f vzero = {};
  v8f acc[8];
  #pragma unroll
  for (int dt = 0; dt < 8; ++dt) acc[dt] = vzero;

  for (int kc = 0; kc < 16; ++kc) {
    const int k0 = kc * 32;
    if (kc + 1 < 16) __builtin_prefetch(aRow + k0 + 32, 0, 0); // global_prefetch_b8

    // A fragment: 4 x b128 f32 loads, 8 x v_perm pack
    AF a;
    {
      const float4 f0 = *(const float4*)(aRow + k0 + khalf);
      const float4 f1 = *(const float4*)(aRow + k0 + khalf + 4);
      const float4 f2 = *(const float4*)(aRow + k0 + khalf + 16);
      const float4 f3 = *(const float4*)(aRow + k0 + khalf + 20);
      a.u[0] = pack_bf16_fast(f0.x, f0.y);
      a.u[1] = pack_bf16_fast(f0.z, f0.w);
      a.u[2] = pack_bf16_fast(f1.x, f1.y);
      a.u[3] = pack_bf16_fast(f1.z, f1.w);
      a.u[4] = pack_bf16_fast(f2.x, f2.y);
      a.u[5] = pack_bf16_fast(f2.z, f2.w);
      a.u[6] = pack_bf16_fast(f3.x, f3.y);
      a.u[7] = pack_bf16_fast(f3.z, f3.w);
    }

    #pragma unroll
    for (int dt = 0; dt < 8; ++dt) {
      AF bm;
      const uint4* bp = (const uint4*)(bBase + dt * 16 * H_ + k0 + khalf);
      bm.u4[0] = bp[0];   // K pairs j=0..3
      bm.u4[1] = bp[2];   // K pairs j=4..7 (+16 ushorts)
      acc[dt] = __builtin_amdgcn_wmma_f32_16x16x32_bf16(
          false, a.v, false, bm.v, (short)0, acc[dt], false, false);
    }
  }

  // conv taps per output row handled by this lane (M = j + 8*hi)
  float lam1[8], la0[8], lap1[8];
  #pragma unroll
  for (int j = 0; j < 8; ++j) {
    const int rl = tg * 16 + 8 * hi + j;  // row within 32-row block
    lam1[j] = sLA[rl]; la0[j] = sLA[rl + 1]; lap1[j] = sLA[rl + 2];
  }

  float sacc[8];
  #pragma unroll
  for (int j = 0; j < 8; ++j) sacc[j] = 0.f;

  #pragma unroll
  for (int dt = 0; dt < 8; ++dt) {
    const int d = d0 + dt * 16 + r;          // C col N = lane%16
    const float qc = qcomb[b * H_ + d];      // q + bias + conv_b
    const float w0 = conv_w[d * 3 + 0];
    const float w1 = conv_w[d * 3 + 1];
    const float w2 = conv_w[d * 3 + 2];
    const float sw = score_w[d];
    #pragma unroll
    for (int j = 0; j < 8; ++j) {
      const float arg = acc[dt][j] + qc + w0 * lam1[j] + w1 * la0[j] + w2 * lap1[j];
      sacc[j] += tanhf(arg) * sw;
    }
  }

  // reduce over the 16 lanes of each half (masks 1,2,4,8 keep bit4 fixed)
  #pragma unroll
  for (int off = 1; off < 16; off <<= 1) {
    #pragma unroll
    for (int j = 0; j < 8; ++j) sacc[j] += __shfl_xor(sacc[j], off, 32);
  }
  if (r == 0) {
    #pragma unroll
    for (int j = 0; j < 8; ++j)
      sPart[(tg * 16 + 8 * hi + j) * 4 + dg] = sacc[j];
  }
  __syncthreads();

  if (threadIdx.x < 32) {   // one wave finishes 32 rows
    const int trow = threadIdx.x;
    const float s = sPart[trow * 4 + 0] + sPart[trow * 4 + 1] +
                    sPart[trow * 4 + 2] + sPart[trow * 4 + 3] + score_b[0];
    const float sc = 1.f / (1.f + __expf(-s));
    scores[b * VL_ + t0b + trow] = sc;
  }
}

// ---------------------------------------------------------------------------
// Kernel 3: per-batch deterministic sum of scores, attn = score/sum,
//           context[b,h] = sum_t attn * value[b,t,h]   (value hits L2, 192MB)
// ---------------------------------------------------------------------------
__global__ __launch_bounds__(256)
void context_kernel(const float* __restrict__ value,
                    const float* __restrict__ scores,
                    float* __restrict__ out_ctx,    // (B,1,H)
                    float* __restrict__ out_attn) { // (B,VL)
  __shared__ float sRed[256];
  __shared__ float sSc[256];
  const int b = blockIdx.x;
  const int tid = threadIdx.x;
  const float* sc = scores + b * VL_;

  float s = 0.f;
  for (int t = tid; t < VL_; t += 256) s += sc[t];
  sRed[tid] = s;
  __syncthreads();
  for (int off = 128; off > 0; off >>= 1) {
    if (tid < off) sRed[tid] += sRed[tid + off];
    __syncthreads();
  }
  const float inv = 1.f / sRed[0];

  float acc0 = 0.f, acc1 = 0.f;
  const float* vb = value + (size_t)b * VL_ * H_;
  for (int tb = 0; tb < VL_; tb += 256) {
    sSc[tid] = sc[tb + tid];
    __syncthreads();
    #pragma unroll 4
    for (int i = 0; i < 256; ++i) {
      const float w = sSc[i];
      const float* row = vb + (size_t)(tb + i) * H_;
      acc0 += w * row[tid];
      acc1 += w * row[tid + 256];
    }
    __syncthreads();
  }
  out_ctx[b * H_ + tid]       = acc0 * inv;
  out_ctx[b * H_ + tid + 256] = acc1 * inv;

  for (int t = tid; t < VL_; t += 256)
    out_attn[b * VL_ + t] = sc[t] * inv;
}

// ---------------------------------------------------------------------------
extern "C" void kernel_launch(void* const* d_in, const int* in_sizes, int n_in,
                              void* d_out, int out_size, void* d_ws, size_t ws_size,
                              hipStream_t stream) {
  const float* query     = (const float*)d_in[0];
  const float* value     = (const float*)d_in[1];
  const float* last_attn = (const float*)d_in[2];
  const float* conv_w    = (const float*)d_in[3];
  const float* conv_b    = (const float*)d_in[4];
  const float* Wq        = (const float*)d_in[5];
  const float* Wv        = (const float*)d_in[6];
  const float* score_w   = (const float*)d_in[7];
  const float* score_b   = (const float*)d_in[8];
  const float* bias      = (const float*)d_in[9];

  float* ws = (float*)d_ws;
  float* qcomb  = ws;                    // 16384 floats
  float* scores = ws + 16384;            // 65536 floats
  unsigned short* wvbf = (unsigned short*)(ws + 16384 + 65536); // 262144 bf16

  float* out_ctx  = (float*)d_out;       // 16384 floats (B,1,H)
  float* out_attn = out_ctx + B_ * H_;   // 65536 floats (B,VL)

  prep_kernel<<<64, 256, 0, stream>>>(query, Wq, Wv, conv_b, bias, qcomb, wvbf);
  score_kernel<<<B_ * (VL_ / 32), 256, 0, stream>>>(
      value, last_attn, conv_w, score_w, score_b, qcomb, wvbf, scores);
  context_kernel<<<B_, 256, 0, stream>>>(value, scores, out_ctx, out_attn);
}
